// DGCNN_4277787427116
// MI455X (gfx1250) — matrix operand order repaired and checked by reading, not compiled, not run on hardware
//
#include <hip/hip_runtime.h>
#include <hip/hip_bf16.h>

typedef __attribute__((ext_vector_type(16))) __bf16 v16bf;
typedef __attribute__((ext_vector_type(8)))  float  v8f;

#define BN_EPS 1e-5f
static constexpr int Bb = 4, Npts = 4096, KNN = 9;

__device__ __forceinline__ float warp_half_sum(float v) {
  // butterfly sum over the 16 lanes of one half-wave (masks < 16 stay in-half)
  v += __shfl_xor(v, 1, 32);
  v += __shfl_xor(v, 2, 32);
  v += __shfl_xor(v, 4, 32);
  v += __shfl_xor(v, 8, 32);
  return v;
}

// ---------------------------------------------------------------------------
// kNN: brute-force per-point top-9 (rank by |xm|^2 - 2*xn.xm; x2[n] is const per row)
// ---------------------------------------------------------------------------
__global__ void knn_kernel(const float* __restrict__ x, int* __restrict__ idxout) {
  __shared__ float xs0[1024], xs1[1024], xs2[1024], x2s[1024];
  const int b = blockIdx.x / (Npts / 256);
  const int n = (blockIdx.x % (Npts / 256)) * 256 + threadIdx.x;
  const float* xb = x + (size_t)b * 3 * Npts;
  const float xn0 = xb[n], xn1 = xb[Npts + n], xn2 = xb[2 * Npts + n];
  float bestd[KNN];
  int   bestm[KNN];
#pragma unroll
  for (int q = 0; q < KNN; ++q) { bestd[q] = 3.0e38f; bestm[q] = 0; }
  for (int chunk = 0; chunk < Npts; chunk += 1024) {
    __syncthreads();
    for (int t = threadIdx.x; t < 1024; t += 256) {
      float v0 = xb[chunk + t], v1 = xb[Npts + chunk + t], v2 = xb[2 * Npts + chunk + t];
      xs0[t] = v0; xs1[t] = v1; xs2[t] = v2;
      x2s[t] = v0 * v0 + v1 * v1 + v2 * v2;
    }
    __syncthreads();
    for (int t = 0; t < 1024; ++t) {
      const int m = chunk + t;
      const float d = x2s[t] - 2.0f * (xn0 * xs0[t] + xn1 * xs1[t] + xn2 * xs2[t]);
      if (m == n) continue;                 // reference adds 1e10*eye -> exclude self
      if (d < bestd[KNN - 1]) {
        int q = KNN - 1;
        while (q > 0 && bestd[q - 1] > d) {
          bestd[q] = bestd[q - 1]; bestm[q] = bestm[q - 1]; --q;
        }
        bestd[q] = d; bestm[q] = m;
      }
    }
  }
  for (int q = 0; q < KNN; ++q) idxout[((size_t)b * Npts + n) * KNN + q] = bestm[q];
}

// ---------------------------------------------------------------------------
// Re-pack fp32 weights (O x Wc) into bf16 WMMA A-fragment order:
//   Wf[((ot*nChunks + ccIdx)*32 + lane)*16 + t]
//   lane: m = lane&15, half = lane>>4 ; element t -> c = cc + half*16 + t
// Each lane of the consumer then loads its whole A fragment as one contiguous
// 32-byte vector load.
// ---------------------------------------------------------------------------
__global__ void pack_weights(const float* __restrict__ W, __bf16* __restrict__ Wf,
                             int Wc, int Ctot, int O) {
  const int nChunks = Ctot >> 5;
  const int total = (O >> 4) * nChunks * 512;
  const int i = blockIdx.x * blockDim.x + threadIdx.x;
  if (i >= total) return;
  const int t = i & 15;
  const int lane = (i >> 4) & 31;
  const int ccIdx = (i >> 9) % nChunks;
  const int ot = (i >> 9) / nChunks;
  const int m = lane & 15, half = lane >> 4;
  const int c = (ccIdx << 5) + half * 16 + t;
  const float v = (c < Wc) ? W[(size_t)(ot * 16 + m) * Wc + c] : 0.f;
  Wf[i] = (__bf16)v;
}

// ---------------------------------------------------------------------------
// Fused (edge-)conv GEMM via WMMA bf16 -> f32.
//   mode 0: accumulate per-channel sum/sumsq of y (pre-BN) -> partial[block][chan][2]
//   mode 1: y*bnA + bnC, max over k, ReLU -> Y
//   mode 2: plain GEMM (KN==1), no BN/ReLU -> Y
// edge=1: channels = [x_i ; x_j - x_i] (2*Cin); edge=0: channels = x (Cin).
// Ctot = channel count padded to a multiple of 32.
// LDS e-tile is stored directly in B-fragment order:
//   eLDS[(((k*nChunks + ccIdx)*32 + half*16 + p) << 4) + t]  (c = cc + half*16 + t)
// so each lane reads its B fragment as one contiguous 32-byte vector load.
// Inner loop double-buffers the B fragment so ds_load latency overlaps the
// previous WMMA instead of serializing behind s_wait_dscnt 0.
// ---------------------------------------------------------------------------
template <int KN>
__global__ void fused_conv(const float* __restrict__ X, const int* __restrict__ idx,
                           const __bf16* __restrict__ Wf, float* __restrict__ Y,
                           float* __restrict__ partial,
                           const float* __restrict__ bnA, const float* __restrict__ bnC,
                           int Cin, int Ctot, int O, int edge, int mode) {
  extern __shared__ char smem[];
  __bf16* eLDS = (__bf16*)smem;
  int*    jLDS = (int*)(smem + (size_t)KN * Ctot * 16 * sizeof(__bf16));

  const int tid = threadIdx.x;
  const int lane = tid & 31, wave = tid >> 5, nwaves = blockDim.x >> 5;
  const int half = lane >> 4, col = lane & 15;
  const int nChunks = Ctot >> 5;
  const int tileN = blockIdx.x * 16;
  const int b = tileN / Npts;
  const int nbase = tileN % Npts;

  // neighbor indices for the 16 points of this tile
  for (int e = tid; e < KN * 16; e += blockDim.x) {
    const int p = e & 15, k = e >> 4;
    jLDS[e] = edge ? idx[((size_t)b * Npts + nbase + p) * KN + k] : 0;
  }
  __syncthreads();

  // stage e-tile for all KN neighbors into swizzled (fragment-order) LDS
  const float* Xb = X + (size_t)b * Cin * Npts;
  for (int e = tid; e < KN * Ctot * 16; e += blockDim.x) {
    const int p = e & 15;            // point within tile (global reads stay coalesced)
    const int rest = e >> 4;
    const int c = rest % Ctot;
    const int k = rest / Ctot;
    const int n = nbase + p;
    float v = 0.f;
    if (c < Cin) {
      v = Xb[(size_t)c * Npts + n];
    } else if (edge && c < 2 * Cin) {
      const int c2 = c - Cin;
      const int j = jLDS[k * 16 + p];
      v = Xb[(size_t)c2 * Npts + j] - Xb[(size_t)c2 * Npts + n];
    }
    const int t = c & 15;
    const int hh = (c >> 4) & 1;
    const int ccIdx = c >> 5;
    eLDS[((((size_t)k * nChunks + ccIdx) * 32 + hh * 16 + p) << 4) + t] = (__bf16)v;
  }
  __syncthreads();

  const v16bf* Wfv = (const v16bf*)Wf;
  const size_t kstride = (size_t)nChunks * 32 * 16;  // elements between k-tiles

  for (int ot = wave; ot * 16 < O; ot += nwaves) {      // wave-uniform: EXEC all-1s at WMMA
    v8f acc[KN];
#pragma unroll
    for (int k = 0; k < KN; ++k) acc[k] = (v8f){};

    __builtin_prefetch(&Wfv[(size_t)ot * nChunks * 32 + lane], 0, 3);  // global_prefetch_b8

    for (int ccIdx = 0; ccIdx < nChunks; ++ccIdx) {
      // A fragment: one 32B vector load from pre-packed weights
      const v16bf afrag = Wfv[((size_t)ot * nChunks + ccIdx) * 32 + lane];
      const __bf16* ebase = eLDS + ((((size_t)ccIdx) * 32 + lane) << 4);
      // B fragments: 32B vector loads from swizzled LDS, double-buffered so the
      // k+1 load is in flight while WMMA k executes.
      v16bf bf_cur = *(const v16bf*)ebase;
#pragma unroll
      for (int k = 0; k < KN; ++k) {
        v16bf bf_nxt = bf_cur;
        if (k + 1 < KN) bf_nxt = *(const v16bf*)(ebase + (size_t)(k + 1) * kstride);
        acc[k] = __builtin_amdgcn_wmma_f32_16x16x32_bf16(
            false, afrag, false, bf_cur, (short)0, acc[k], false, false);
        bf_cur = bf_nxt;
      }
    }

    if (mode == 0) {
#pragma unroll
      for (int r = 0; r < 8; ++r) {
        float s = 0.f, q = 0.f;
#pragma unroll
        for (int k = 0; k < KN; ++k) { const float y = acc[k][r]; s += y; q += y * y; }
        s = warp_half_sum(s);
        q = warp_half_sum(q);
        if (col == 0) {
          const int m = ot * 16 + r + 8 * half;     // C/D row: M = r + 8*half
          partial[(size_t)blockIdx.x * (O * 2) + m * 2 + 0] = s;
          partial[(size_t)blockIdx.x * (O * 2) + m * 2 + 1] = q;
        }
      }
    } else if (mode == 1) {
#pragma unroll
      for (int r = 0; r < 8; ++r) {
        const int m = ot * 16 + r + 8 * half;
        const float a = bnA[m], c0 = bnC[m];
        float v = -3.0e38f;
#pragma unroll
        for (int k = 0; k < KN; ++k) v = fmaxf(v, acc[k][r] * a + c0);
        v = fmaxf(v, 0.f);                          // relu(max(bn(y))) == max(relu(bn(y)))
        Y[((size_t)b * O + m) * Npts + nbase + col] = v;
      }
    } else {
#pragma unroll
      for (int r = 0; r < 8; ++r) {
        const int m = ot * 16 + r + 8 * half;
        Y[((size_t)b * O + m) * Npts + nbase + col] = acc[0][r];
      }
    }
  }
}

// ---------------------------------------------------------------------------
// Deterministic stats finalize: serial in-order sum over block partials.
// ---------------------------------------------------------------------------
__global__ void finalize_stats(const float* __restrict__ partial, int nblocks, int O,
                               const float* __restrict__ g, const float* __restrict__ beta,
                               float invCnt, float* __restrict__ bnA, float* __restrict__ bnC) {
  const int m = blockIdx.x * blockDim.x + threadIdx.x;
  if (m >= O) return;
  float s = 0.f, q = 0.f;
  for (int t = 0; t < nblocks; ++t) {
    s += partial[(size_t)t * (O * 2) + m * 2 + 0];
    q += partial[(size_t)t * (O * 2) + m * 2 + 1];
  }
  const float mean = s * invCnt;
  const float var = fmaxf(q * invCnt - mean * mean, 0.f);
  const float a = g[m] * rsqrtf(var + BN_EPS);
  bnA[m] = a;
  bnC[m] = beta[m] - mean * a;
}

__global__ void concat_kernel(const float* __restrict__ out4, const float* __restrict__ low,
                              float* __restrict__ cat) {
  const int i = blockIdx.x * blockDim.x + threadIdx.x;
  const int total = Bb * 304 * Npts;
  if (i >= total) return;
  const int n = i % Npts;
  const int rest = i / Npts;
  const int c = rest % 304;
  const int b = rest / 304;
  cat[i] = (c < 256) ? out4[((size_t)b * 256 + c) * Npts + n]
                     : low[((size_t)b * 48 + (c - 256)) * Npts + n];
}

__global__ void reduce_maxmean(const float* __restrict__ fused, float* __restrict__ h) {
  __shared__ float smax[256], ssum[256];
  const int b = blockIdx.x / 256, o = blockIdx.x % 256;
  const float* row = fused + ((size_t)b * 256 + o) * Npts;
  float mx = -3.0e38f, sm = 0.f;
  for (int n = threadIdx.x; n < Npts; n += 256) {
    const float v = row[n];
    mx = fmaxf(mx, v);
    sm += v;
  }
  smax[threadIdx.x] = mx; ssum[threadIdx.x] = sm;
  __syncthreads();
  for (int st = 128; st > 0; st >>= 1) {
    if (threadIdx.x < st) {
      smax[threadIdx.x] = fmaxf(smax[threadIdx.x], smax[threadIdx.x + st]);
      ssum[threadIdx.x] += ssum[threadIdx.x + st];
    }
    __syncthreads();
  }
  if (threadIdx.x == 0) {
    h[b * 512 + o] = smax[0];
    h[b * 512 + 256 + o] = ssum[0] * (1.0f / Npts);
  }
}

__global__ void classifier_tail(const float* __restrict__ h, const float* __restrict__ w1,
                                const float* __restrict__ g, const float* __restrict__ beta,
                                const float* __restrict__ w2, float* __restrict__ out) {
  __shared__ float hs[Bb * 512];
  __shared__ float h2[Bb * 256];
  for (int i = threadIdx.x; i < Bb * 512; i += blockDim.x) hs[i] = h[i];
  __syncthreads();
  const int o = threadIdx.x;  // 256 threads, one per channel
  float y[Bb];
  float s = 0.f, q = 0.f;
  const float* wr = w1 + (size_t)o * 512;
#pragma unroll
  for (int b = 0; b < Bb; ++b) {
    float acc = 0.f;
    for (int c = 0; c < 512; ++c) acc += wr[c] * hs[b * 512 + c];
    y[b] = acc; s += acc; q += acc * acc;
  }
  const float mean = s * (1.0f / Bb);
  const float var = fmaxf(q * (1.0f / Bb) - mean * mean, 0.f);
  const float a = g[o] * rsqrtf(var + BN_EPS);
  const float c0 = beta[o] - mean * a;
#pragma unroll
  for (int b = 0; b < Bb; ++b) h2[b * 256 + o] = fmaxf(y[b] * a + c0, 0.f);
  __syncthreads();
  if (threadIdx.x < Bb * 40) {
    const int b = threadIdx.x / 40, o2 = threadIdx.x % 40;
    float acc = 0.f;
    const float* w2r = w2 + (size_t)o2 * 256;
    for (int c = 0; c < 256; ++c) acc += w2r[c] * h2[b * 256 + c];
    out[b * 40 + o2] = acc;
  }
}

// ---------------------------------------------------------------------------
extern "C" void kernel_launch(void* const* d_in, const int* in_sizes, int n_in,
                              void* d_out, int out_size, void* d_ws, size_t ws_size,
                              hipStream_t stream) {
  (void)in_sizes; (void)n_in; (void)out_size; (void)ws_size;
  const float* x   = (const float*)d_in[0];
  const float* w1  = (const float*)d_in[2];
  const float* g1  = (const float*)d_in[3];
  const float* b1  = (const float*)d_in[4];
  const float* w2  = (const float*)d_in[5];
  const float* g2  = (const float*)d_in[6];
  const float* b2  = (const float*)d_in[7];
  const float* w3  = (const float*)d_in[8];
  const float* g3  = (const float*)d_in[9];
  const float* b3  = (const float*)d_in[10];
  const float* w4  = (const float*)d_in[11];
  const float* g4  = (const float*)d_in[12];
  const float* b4  = (const float*)d_in[13];
  const float* dw  = (const float*)d_in[14];
  const float* dg  = (const float*)d_in[15];
  const float* db  = (const float*)d_in[16];
  const float* fw  = (const float*)d_in[17];
  const float* cw1 = (const float*)d_in[18];
  const float* cg  = (const float*)d_in[19];
  const float* cb  = (const float*)d_in[20];
  const float* cw2 = (const float*)d_in[21];
  float* outp = (float*)d_out;

  float* ws = (float*)d_ws;
  size_t off = 0;
  int*    idxb    = (int*)(ws + off);    off += (size_t)Bb * Npts * KNN;
  float*  out1    = ws + off;            off += (size_t)Bb * 64 * Npts;
  float*  bufA    = ws + off;            off += (size_t)Bb * 256 * Npts;
  float*  bufB    = ws + off;            off += (size_t)Bb * 256 * Npts;
  float*  low     = ws + off;            off += (size_t)Bb * 48 * Npts;
  float*  cat     = ws + off;            off += (size_t)Bb * 304 * Npts;
  float*  partial = ws + off;            off += (size_t)(Bb * Npts / 16) * 256 * 2;
  float*  bnA     = ws + off;            off += 512;
  float*  bnC     = ws + off;            off += 512;
  float*  hbuf    = ws + off;            off += Bb * 512;  // 2048 (keeps 32B alignment)
  __bf16* wfrag   = (__bf16*)(ws + off); off += 65536;     // max 131072 bf16 (layer 4)
  float*  fused   = bufB;  // bufB free after layer 4 reads it

  const int tiles = Bb * Npts / 16;  // 1024 point-tile blocks
  const float invEC = 1.0f / (float)(Bb * Npts * KNN);  // edgeconv BN count
  const float invPC = 1.0f / (float)(Bb * Npts);        // pointwise BN count

  // allow >48KB dynamic LDS (layer 4 needs 9*512*16*2B ~= 144KB; WGP has 320KB)
  const int sh9max = (int)((size_t)KNN * 512 * 16 * sizeof(__bf16) + KNN * 16 * sizeof(int));
  const int sh1max = (int)((size_t)320 * 16 * sizeof(__bf16) + 16 * sizeof(int));
  hipFuncSetAttribute(reinterpret_cast<const void*>(fused_conv<KNN>),
                      hipFuncAttributeMaxDynamicSharedMemorySize, sh9max);
  hipFuncSetAttribute(reinterpret_cast<const void*>(fused_conv<1>),
                      hipFuncAttributeMaxDynamicSharedMemorySize, sh1max);

  knn_kernel<<<Bb * (Npts / 256), 256, 0, stream>>>(x, idxb);

  auto packGrid = [](int O, int Ctot) { return ((O >> 4) * (Ctot >> 5) * 512 + 255) / 256; };

  // Layer 1: Cin=3 (pad Ctot=32), Wc=6, O=64
  {
    const size_t sh = (size_t)KNN * 32 * 16 * sizeof(__bf16) + KNN * 16 * sizeof(int);
    pack_weights<<<packGrid(64, 32), 256, 0, stream>>>(w1, wfrag, 6, 32, 64);
    fused_conv<KNN><<<tiles, 128, sh, stream>>>(x, idxb, wfrag, nullptr, partial, nullptr, nullptr,
                                                3, 32, 64, 1, 0);
    finalize_stats<<<1, 256, 0, stream>>>(partial, tiles, 64, g1, b1, invEC, bnA, bnC);
    fused_conv<KNN><<<tiles, 128, sh, stream>>>(x, idxb, wfrag, out1, nullptr, bnA, bnC,
                                                3, 32, 64, 1, 1);
  }
  // Layer 2: Cin=64, Ctot=128, O=128
  {
    const size_t sh = (size_t)KNN * 128 * 16 * sizeof(__bf16) + KNN * 16 * sizeof(int);
    pack_weights<<<packGrid(128, 128), 256, 0, stream>>>(w2, wfrag, 128, 128, 128);
    fused_conv<KNN><<<tiles, 128, sh, stream>>>(out1, idxb, wfrag, nullptr, partial, nullptr,
                                                nullptr, 64, 128, 128, 1, 0);
    finalize_stats<<<1, 256, 0, stream>>>(partial, tiles, 128, g2, b2, invEC, bnA, bnC);
    fused_conv<KNN><<<tiles, 128, sh, stream>>>(out1, idxb, wfrag, bufA, nullptr, bnA, bnC,
                                                64, 128, 128, 1, 1);
  }
  // Layer 3: Cin=128, Ctot=256, O=256
  {
    const size_t sh = (size_t)KNN * 256 * 16 * sizeof(__bf16) + KNN * 16 * sizeof(int);
    pack_weights<<<packGrid(256, 256), 256, 0, stream>>>(w3, wfrag, 256, 256, 256);
    fused_conv<KNN><<<tiles, 128, sh, stream>>>(bufA, idxb, wfrag, nullptr, partial, nullptr,
                                                nullptr, 128, 256, 256, 1, 0);
    finalize_stats<<<1, 256, 0, stream>>>(partial, tiles, 256, g3, b3, invEC, bnA, bnC);
    fused_conv<KNN><<<tiles, 128, sh, stream>>>(bufA, idxb, wfrag, bufB, nullptr, bnA, bnC,
                                                128, 256, 256, 1, 1);
  }
  // Layer 4: Cin=256, Ctot=512, O=256
  {
    const size_t sh = (size_t)KNN * 512 * 16 * sizeof(__bf16) + KNN * 16 * sizeof(int);
    pack_weights<<<packGrid(256, 512), 256, 0, stream>>>(w4, wfrag, 512, 512, 256);
    fused_conv<KNN><<<tiles, 128, sh, stream>>>(bufB, idxb, wfrag, nullptr, partial, nullptr,
                                                nullptr, 256, 512, 256, 1, 0);
    finalize_stats<<<1, 256, 0, stream>>>(partial, tiles, 256, g4, b4, invEC, bnA, bnC);
    fused_conv<KNN><<<tiles, 128, sh, stream>>>(bufB, idxb, wfrag, bufA, nullptr, bnA, bnC,
                                                256, 512, 256, 1, 1);
  }
  // Decoder branch: pointwise 48x64 GEMM + BN + relu on out1
  {
    const size_t sh = (size_t)64 * 16 * sizeof(__bf16) + 16 * sizeof(int);
    pack_weights<<<packGrid(48, 64), 256, 0, stream>>>(dw, wfrag, 64, 64, 48);
    fused_conv<1><<<tiles, 128, sh, stream>>>(out1, nullptr, wfrag, nullptr, partial, nullptr,
                                              nullptr, 64, 64, 48, 0, 0);
    finalize_stats<<<1, 256, 0, stream>>>(partial, tiles, 48, dg, db, invPC, bnA, bnC);
    fused_conv<1><<<tiles, 128, sh, stream>>>(out1, nullptr, wfrag, low, nullptr, bnA, bnC,
                                              64, 64, 48, 0, 1);
  }
  // Fusion: concat(out4, low) -> 304 ch, plain GEMM 256x304 (Ctot pad 320)
  concat_kernel<<<(Bb * 304 * Npts + 255) / 256, 256, 0, stream>>>(bufA, low, cat);
  {
    const size_t sh = (size_t)320 * 16 * sizeof(__bf16) + 16 * sizeof(int);
    pack_weights<<<packGrid(256, 320), 256, 0, stream>>>(fw, wfrag, 304, 320, 256);
    fused_conv<1><<<tiles, 128, sh, stream>>>(cat, nullptr, wfrag, fused, nullptr, nullptr,
                                              nullptr, 304, 320, 256, 0, 2);
  }
  // Global max/mean pool -> h (B,512)
  reduce_maxmean<<<Bb * 256, 256, 0, stream>>>(fused, hbuf);
  // Classifier head -> logits (B,40)
  classifier_tail<<<1, 256, 0, stream>>>(hbuf, cw1, cg, cb, cw2, outp);
}
